// Bottleneck_1w1a_24386824307039
// MI455X (gfx1250) — 8-bit matrix operand order repaired and checked by reading, compiled, not run on hardware
//
#include <hip/hip_runtime.h>
#include <hip/hip_bf16.h>


typedef __attribute__((ext_vector_type(8))) int v8i_t;

#define NB     64
#define CIN    256
#define PL     64
#define HH     56
#define WW     56
#define HWSZ   (HH * WW)
#define NPIX   (NB * HWSZ)
#define HP     58
#define EPSBN  1e-5f

#define XB_OFF     ((size_t)0)
#define XB_BYTES   ((size_t)NPIX * 256)
#define RAW1_OFF   (XB_OFF + XB_BYTES)
#define RAW1_BYTES ((size_t)NPIX * 64 * 4)
#define A1P_OFF    (RAW1_OFF + RAW1_BYTES)
#define A1P_BYTES  ((size_t)NB * HP * HP * 64)
#define RAW2_OFF   (A1P_OFF + A1P_BYTES)
#define RAW2_BYTES ((size_t)NPIX * 64 * 4)
#define A2_OFF     (RAW2_OFF + RAW2_BYTES)
#define A2_BYTES   ((size_t)NPIX * 64)
#define RAW3_OFF   (A2_OFF + A2_BYTES)
#define RAW3_BYTES ((size_t)NPIX * 256 * 4)
#define W1B_OFF    (RAW3_OFF + RAW3_BYTES)
#define W2B_OFF    (W1B_OFF + (size_t)64 * 256)
#define W3B_OFF    (W2B_OFF + (size_t)64 * 576)
#define STATS_OFF  (W3B_OFF + (size_t)256 * 64)
#define STATS_U64  (6 * 256)
#define STATS_F32  (6 * 256)
#define STATS_BYTES ((size_t)STATS_U64 * 8 + (size_t)STATS_F32 * 4)

__device__ __forceinline__ signed char sgn8(float v) {
    return (v > 0.0f) ? (signed char)1 : ((v < 0.0f) ? (signed char)-1 : (signed char)0);
}

__device__ __forceinline__ v8i_t load_b_frag(const signed char* row64, int khalf) {
    const int4 lo = *(const int4*)(row64 + khalf * 16);
    const int4 hi = *(const int4*)(row64 + khalf * 16 + 32);
    v8i_t r;
    r[0] = lo.x; r[1] = lo.y; r[2] = lo.z; r[3] = lo.w;
    r[4] = hi.x; r[5] = hi.y; r[6] = hi.z; r[7] = hi.w;
    return r;
}

__device__ __forceinline__ v8i_t load_a_frag(const signed char* rowK, int khalf) {
    v8i_t a;
#pragma unroll
    for (int j = 0; j < 2; ++j) {
        const int4 d = *(const int4*)(rowK + khalf * 16 + j * 32);
        a[4 * j]     = d.x;
        a[4 * j + 1] = d.y;
        a[4 * j + 2] = d.z;
        a[4 * j + 3] = d.w;
    }
    return a;
}

__device__ __forceinline__ void store_c_tile(int* dst8, const v8i_t c) {
    int4 lo, hi;
    lo.x = c[0]; lo.y = c[1]; lo.z = c[2]; lo.w = c[3];
    hi.x = c[4]; hi.y = c[5]; hi.z = c[6]; hi.w = c[7];
    *(int4*)dst8       = lo;
    *(int4*)(dst8 + 4) = hi;
}


__global__ __launch_bounds__(256) void pack_x_kernel(const float* __restrict__ x,
                                                     signed char* __restrict__ xb) {
    const size_t total = (size_t)NB * CIN * HWSZ;
    for (size_t idx = (size_t)blockIdx.x * blockDim.x + threadIdx.x; idx < total;
         idx += (size_t)gridDim.x * blockDim.x) {
        const size_t hw = idx % HWSZ;
        const size_t c  = (idx / HWSZ) % CIN;
        const size_t n  = idx / ((size_t)HWSZ * CIN);
        xb[(n * HWSZ + hw) * CIN + c] = sgn8(x[idx]);
    }
}

__global__ __launch_bounds__(256) void sign_copy_kernel(const float* __restrict__ src,
                                                        signed char* __restrict__ dst, int n) {
    const int i = blockIdx.x * blockDim.x + threadIdx.x;
    if (i < n) dst[i] = sgn8(src[i]);
}

__global__ __launch_bounds__(256) void pack_w2_kernel(const float* __restrict__ w,
                                                      signed char* __restrict__ wb) {
    const int i = blockIdx.x * blockDim.x + threadIdx.x;
    if (i < 64 * 64 * 9) {
        const int oc  = i / 576;
        const int rem = i % 576;
        const int ic  = rem / 9;
        const int tap = rem % 9;
        wb[oc * 576 + tap * 64 + ic] = sgn8(w[i]);
    }
}

__global__ __launch_bounds__(128) void gemm1_kernel(const signed char* __restrict__ xb,
                                                    const signed char* __restrict__ w1b,
                                                    int* __restrict__ raw1) {
    __shared__ __align__(16) signed char Ws[64 * 256];
    __shared__ __align__(16) signed char Bt[64 * 64];

    const int tid = threadIdx.x;
    for (int i = tid; i < 64 * 256 / 16; i += 128)
        ((int4*)Ws)[i] = ((const int4*)w1b)[i];

    const int lane  = tid & 31;
    const int wave  = tid >> 5;
    const int col   = lane & 15;
    const int khalf = lane >> 4;
    const int blkPix = blockIdx.x * 64;

    v8i_t acc[4] = {};

#pragma unroll
    for (int ks = 0; ks < 256; ks += 64) {
        __syncthreads();
        for (int i = tid; i < 64 * 4; i += 128) {
            const int p = i >> 2, q = i & 3;
            ((int4*)Bt)[i] =
                *(const int4*)(xb + (size_t)(blkPix + p) * 256 + ks + q * 16);
        }
        if (ks < 192)
            __builtin_prefetch(xb + (size_t)(blkPix + tid) * 2 + ks + 64, 0, 0);
        __syncthreads();

        const v8i_t bfrag = load_b_frag(Bt + (wave * 16 + col) * 64, khalf);
#pragma unroll
        for (int mt = 0; mt < 4; ++mt) {
            const v8i_t afrag = load_a_frag(Ws + (mt * 16 + col) * 256 + ks, khalf);
            acc[mt] = __builtin_amdgcn_wmma_i32_16x16x64_iu8(true, afrag, true, bfrag,
                                                             acc[mt], false, false);
        }
    }

    const int pix = blkPix + wave * 16 + col;
#pragma unroll
    for (int mt = 0; mt < 4; ++mt)
        store_c_tile(raw1 + (size_t)pix * 64 + mt * 16 + khalf * 8, acc[mt]);
}

__global__ __launch_bounds__(128) void gemm2_kernel(const signed char* __restrict__ a1p,
                                                    const signed char* __restrict__ w2b,
                                                    int* __restrict__ raw2) {
    __shared__ __align__(16) signed char Ws[64 * 576];
    __shared__ __align__(16) signed char Bt[64 * 64];

    const int tid = threadIdx.x;
    for (int i = tid; i < 64 * 576 / 16; i += 128)
        ((int4*)Ws)[i] = ((const int4*)w2b)[i];

    const int lane  = tid & 31;
    const int wave  = tid >> 5;
    const int col   = lane & 15;
    const int khalf = lane >> 4;
    const int blkPix = blockIdx.x * 64;

    v8i_t acc[4] = {};

    for (int tap = 0; tap < 9; ++tap) {
        const int dy = tap / 3, dx = tap % 3;
        __syncthreads();
        for (int i = tid; i < 64 * 4; i += 128) {
            const int p  = i >> 2, q = i & 3;
            const int pix = blkPix + p;
            const int n  = pix / HWSZ;
            const int hw = pix % HWSZ;
            const int y  = hw / WW, x = hw % WW;
            const signed char* src =
                a1p + (((size_t)n * HP + (y + dy)) * HP + (x + dx)) * 64 + q * 16;
            ((int4*)Bt)[i] = *(const int4*)src;
        }
        __syncthreads();

        const v8i_t bfrag = load_b_frag(Bt + (wave * 16 + col) * 64, khalf);
#pragma unroll
        for (int mt = 0; mt < 4; ++mt) {
            const v8i_t afrag = load_a_frag(Ws + (mt * 16 + col) * 576 + tap * 64, khalf);
            acc[mt] = __builtin_amdgcn_wmma_i32_16x16x64_iu8(true, afrag, true, bfrag,
                                                             acc[mt], false, false);
        }
    }

    const int pix = blkPix + wave * 16 + col;
#pragma unroll
    for (int mt = 0; mt < 4; ++mt)
        store_c_tile(raw2 + (size_t)pix * 64 + mt * 16 + khalf * 8, acc[mt]);
}

__global__ __launch_bounds__(128) void gemm3_kernel(const signed char* __restrict__ a2,
                                                    const signed char* __restrict__ w3b,
                                                    int* __restrict__ raw3) {
    __shared__ __align__(16) signed char Ws[256 * 64];
    __shared__ __align__(16) signed char Bt[16 * 64];

    const int tid = threadIdx.x;
    for (int i = tid; i < 256 * 64 / 16; i += 128)
        ((int4*)Ws)[i] = ((const int4*)w3b)[i];

    const int lane  = tid & 31;
    const int wave  = tid >> 5;
    const int col   = lane & 15;
    const int khalf = lane >> 4;
    const int blkPix = blockIdx.x * 16;

    for (int i = tid; i < 16 * 4; i += 128) {
        const int p = i >> 2, q = i & 3;
        ((int4*)Bt)[i] = *(const int4*)(a2 + (size_t)(blkPix + p) * 64 + q * 16);
    }
    __syncthreads();

    const v8i_t bfrag = load_b_frag(Bt + col * 64, khalf);

    v8i_t acc[4] = {};
#pragma unroll
    for (int mt = 0; mt < 4; ++mt) {
        const int m0 = wave * 64 + mt * 16;
        const v8i_t afrag = load_a_frag(Ws + (m0 + col) * 64, khalf);
        acc[mt] = __builtin_amdgcn_wmma_i32_16x16x64_iu8(true, afrag, true, bfrag,
                                                         acc[mt], false, false);
    }

    const int pix = blkPix + col;
#pragma unroll
    for (int mt = 0; mt < 4; ++mt)
        store_c_tile(raw3 + (size_t)pix * 256 + wave * 64 + mt * 16 + khalf * 8, acc[mt]);
}

__global__ __launch_bounds__(256) void stats_kernel(const int* __restrict__ raw, int pch,
                                                    long long n_elems,
                                                    unsigned long long* __restrict__ sum,
                                                    unsigned long long* __restrict__ sumsq) {
    __shared__ unsigned long long ss[256];
    __shared__ unsigned long long sq[256];
    for (int i = threadIdx.x; i < pch; i += blockDim.x) { ss[i] = 0ull; sq[i] = 0ull; }
    __syncthreads();
    const long long stride = (long long)gridDim.x * blockDim.x;
    for (long long idx = (long long)blockIdx.x * blockDim.x + threadIdx.x; idx < n_elems;
         idx += stride) {
        const long long v  = raw[idx];
        const int       ch = (int)(idx & (pch - 1));
        atomicAdd(&ss[ch], (unsigned long long)v);
        atomicAdd(&sq[ch], (unsigned long long)(v * v));
    }
    __syncthreads();
    for (int i = threadIdx.x; i < pch; i += blockDim.x) {
        atomicAdd(&sum[i], ss[i]);
        atomicAdd(&sumsq[i], sq[i]);
    }
}

__global__ __launch_bounds__(256) void finalize_stats_kernel(
    const unsigned long long* __restrict__ sum, const unsigned long long* __restrict__ sumsq,
    const float* __restrict__ g, const float* __restrict__ b, int pch, double inv_n,
    float* __restrict__ scale, float* __restrict__ shift) {
    const int c = blockIdx.x * blockDim.x + threadIdx.x;
    if (c < pch) {
        const double m  = (double)(long long)sum[c] * inv_n;
        const double v  = (double)(long long)sumsq[c] * inv_n - m * m;
        const float  sc = g[c] * rsqrtf((float)v + EPSBN);
        scale[c] = sc;
        shift[c] = b[c] - (float)m * sc;
    }
}

__global__ __launch_bounds__(256) void bnact_pad_kernel(const int* __restrict__ raw,
                                                        const float* __restrict__ sc,
                                                        const float* __restrict__ sh,
                                                        signed char* __restrict__ a1p) {
    const size_t total = (size_t)NPIX * 64;
    for (size_t idx = (size_t)blockIdx.x * blockDim.x + threadIdx.x; idx < total;
         idx += (size_t)gridDim.x * blockDim.x) {
        const int    p   = (int)(idx & 63);
        const size_t pix = idx >> 6;
        const int    n   = (int)(pix / HWSZ);
        const int    hw  = (int)(pix % HWSZ);
        const int    y   = hw / WW, x = hw % WW;
        const float  t   = sc[p] * (float)raw[idx] + sh[p];
        a1p[(((size_t)n * HP + (y + 1)) * HP + (x + 1)) * 64 + p] = sgn8(t);
    }
}

__global__ __launch_bounds__(256) void bnact_kernel(const int* __restrict__ raw,
                                                    const float* __restrict__ sc,
                                                    const float* __restrict__ sh,
                                                    signed char* __restrict__ a2) {
    const size_t total = (size_t)NPIX * 64;
    for (size_t idx = (size_t)blockIdx.x * blockDim.x + threadIdx.x; idx < total;
         idx += (size_t)gridDim.x * blockDim.x) {
        const int   p = (int)(idx & 63);
        const float t = sc[p] * (float)raw[idx] + sh[p];
        a2[idx] = sgn8(t);
    }
}

__global__ __launch_bounds__(256) void final_kernel(const int* __restrict__ raw3,
                                                    const float* __restrict__ sc,
                                                    const float* __restrict__ sh,
                                                    const float* __restrict__ x,
                                                    float* __restrict__ out) {
    const size_t total = (size_t)NB * CIN * HWSZ;
    for (size_t idx = (size_t)blockIdx.x * blockDim.x + threadIdx.x; idx < total;
         idx += (size_t)gridDim.x * blockDim.x) {
        const size_t hw = idx % HWSZ;
        const size_t c  = (idx / HWSZ) % CIN;
        const size_t n  = idx / ((size_t)HWSZ * CIN);
        const size_t pix = n * HWSZ + hw;
        float v = sc[c] * (float)raw3[pix * 256 + c] + sh[c] + x[idx];
        v = fminf(1.0f, fmaxf(-1.0f, v));
        out[idx] = v;
    }
}

extern "C" void kernel_launch(void* const* d_in, const int* in_sizes, int n_in,
                              void* d_out, int out_size, void* d_ws, size_t ws_size,
                              hipStream_t stream) {
    const float* x  = (const float*)d_in[0];
    const float* w1 = (const float*)d_in[1];
    const float* w2 = (const float*)d_in[2];
    const float* w3 = (const float*)d_in[3];
    const float* g1 = (const float*)d_in[4];
    const float* b1 = (const float*)d_in[5];
    const float* g2 = (const float*)d_in[6];
    const float* b2 = (const float*)d_in[7];
    const float* g3 = (const float*)d_in[8];
    const float* b3 = (const float*)d_in[9];

    char* ws = (char*)d_ws;
    signed char* xb   = (signed char*)(ws + XB_OFF);
    int*         raw1 = (int*)(ws + RAW1_OFF);
    signed char* a1p  = (signed char*)(ws + A1P_OFF);
    int*         raw2 = (int*)(ws + RAW2_OFF);
    signed char* a2   = (signed char*)(ws + A2_OFF);
    int*         raw3 = (int*)(ws + RAW3_OFF);
    signed char* w1b  = (signed char*)(ws + W1B_OFF);
    signed char* w2b  = (signed char*)(ws + W2B_OFF);
    signed char* w3b  = (signed char*)(ws + W3B_OFF);

    unsigned long long* u64s = (unsigned long long*)(ws + STATS_OFF);
    unsigned long long *sum1 = u64s + 0,   *sq1 = u64s + 256;
    unsigned long long *sum2 = u64s + 512, *sq2 = u64s + 768;
    unsigned long long *sum3 = u64s + 1024,*sq3 = u64s + 1280;
    float* f32s = (float*)(ws + STATS_OFF + (size_t)STATS_U64 * 8);
    float *sc1 = f32s + 0,    *sh1 = f32s + 256;
    float *sc2 = f32s + 512,  *sh2 = f32s + 768;
    float *sc3 = f32s + 1024, *sh3 = f32s + 1280;

    const double invN = 1.0 / (double)NPIX;

    hipMemsetAsync(a1p, 0, A1P_BYTES, stream);
    hipMemsetAsync(u64s, 0, STATS_BYTES, stream);

    pack_x_kernel<<<4096, 256, 0, stream>>>(x, xb);
    sign_copy_kernel<<<(64 * 256 + 255) / 256, 256, 0, stream>>>(w1, w1b, 64 * 256);
    pack_w2_kernel<<<(64 * 64 * 9 + 255) / 256, 256, 0, stream>>>(w2, w2b);
    sign_copy_kernel<<<(256 * 64 + 255) / 256, 256, 0, stream>>>(w3, w3b, 256 * 64);

    gemm1_kernel<<<NPIX / 64, 128, 0, stream>>>(xb, w1b, raw1);
    stats_kernel<<<1024, 256, 0, stream>>>(raw1, 64, (long long)NPIX * 64, sum1, sq1);
    finalize_stats_kernel<<<1, 256, 0, stream>>>(sum1, sq1, g1, b1, 64, invN, sc1, sh1);
    bnact_pad_kernel<<<4096, 256, 0, stream>>>(raw1, sc1, sh1, a1p);

    gemm2_kernel<<<NPIX / 64, 128, 0, stream>>>(a1p, w2b, raw2);
    stats_kernel<<<1024, 256, 0, stream>>>(raw2, 64, (long long)NPIX * 64, sum2, sq2);
    finalize_stats_kernel<<<1, 256, 0, stream>>>(sum2, sq2, g2, b2, 64, invN, sc2, sh2);
    bnact_kernel<<<4096, 256, 0, stream>>>(raw2, sc2, sh2, a2);

    gemm3_kernel<<<NPIX / 16, 128, 0, stream>>>(a2, w3b, raw3);
    stats_kernel<<<2048, 256, 0, stream>>>(raw3, 256, (long long)NPIX * 256, sum3, sq3);
    finalize_stats_kernel<<<1, 256, 0, stream>>>(sum3, sq3, g3, b3, 256, invN, sc3, sh3);
    final_kernel<<<8192, 256, 0, stream>>>(raw3, sc3, sh3, x, (float*)d_out);
}
